// MultiHeadAttModel_allin_52536039964914
// MI455X (gfx1250) — matrix operand order repaired and checked by reading, not compiled, not run on hardware
//
#include <hip/hip_runtime.h>
#include <math.h>

#define B_SZ    32
#define N_AG    64
#define E_DIM   256
#define DV_     32
#define NV_     8
#define HIS_    50
#define DK_     9
#define M_ROWS  2048   // B_SZ * N_AG

// ---- WMMA types (gfx1250, wave32) ----
typedef __attribute__((ext_vector_type(16))) __bf16          v16bf;
typedef __attribute__((ext_vector_type(8)))  float           v8f;
typedef __attribute__((ext_vector_type(8)))  unsigned short  v8us;
typedef __attribute__((ext_vector_type(16))) unsigned short  v16us;

union BF16Vec { v16us u; v16bf b; };

__device__ __forceinline__ unsigned short f2bf(float f) {
  unsigned int u = __float_as_uint(f);
  u += 0x7FFFu + ((u >> 16) & 1u);          // round-to-nearest-even
  return (unsigned short)(u >> 16);
}

// ------------------------------------------------------------------
// Prep: pack weights to WMMA-B bf16 layout, adjacency argmax, PE table
// Packed layout per (kt,ct) tile: lane l holds 16 contiguous halfs =
// W[k][n], k = kt*32 + (l>=16 ? 8 : 0) + (i<8 ? i : 8+i) (i=0..15), n = ct*16 + (l&15).
// ------------------------------------------------------------------
__global__ __launch_bounds__(256) void prep_kernel(
    const float* __restrict__ Wq, const float* __restrict__ Wk,
    const float* __restrict__ Wv, const float* __restrict__ Wo,
    const float* __restrict__ adjs,
    unsigned short* __restrict__ wpack,   // 4 * 65536 ushorts
    int* __restrict__ idx,                // [N_AG*HIS_]
    float* __restrict__ pe) {             // [HIS_*256]
  int g = blockIdx.x * 256 + threadIdx.x;
  if (g < 262144) {
    int m    = g >> 16;
    int r    = g & 65535;
    int i    = r & 15;
    int lane = (r >> 4) & 31;
    int tl   = r >> 9;                     // 0..127
    int ct   = tl & 15, kt = tl >> 4;
    int kk   = kt * 32 + ((lane >> 4) << 3) + (i < 8 ? i : i + 8);
    int nn   = ct * 16 + (lane & 15);
    const float* W = (m == 0) ? Wq : (m == 1) ? Wk : (m == 2) ? Wv : Wo;
    wpack[g] = f2bf(W[kk * 256 + nn]);
  } else if (g < 262144 + N_AG * HIS_) {
    int r = g - 262144;                    // n*HIS + t
    const float* row = adjs + r * N_AG;
    int a = 0;
    for (int A = 0; A < N_AG; ++A) { if (row[A] > 0.5f) { a = A; break; } }
    idx[r] = a;
  } else if (g < 262144 + N_AG * HIS_ + HIS_ * 256) {
    int r = g - 262144 - N_AG * HIS_;
    int t = r >> 8, e = r & 255;
    float div = __expf(-logf(10000.0f) / 256.0f * (float)(e & ~1));
    float ang = (float)t * div;
    pe[r] = (e & 1) ? cosf(ang) : sinf(ang);
  }
}

// ------------------------------------------------------------------
// Generic M=2048 GEMM: Out = relu(A(f32)[2048,256] @ Wp(bf16 packed) + bias)
// grid = 64 blocks * 8 waves; wave -> (row tile rt, 4 col tiles cg*4..)
// ------------------------------------------------------------------
__global__ __launch_bounds__(256) void gemm256_relu_kernel(
    const float* __restrict__ A, const unsigned short* __restrict__ Wp,
    const float* __restrict__ bias, float* __restrict__ Out) {
  const int w     = threadIdx.x >> 5;
  const int lane  = threadIdx.x & 31;
  const int wid   = blockIdx.x * 8 + w;
  const int rt    = wid >> 2;              // 0..127
  const int cg    = wid & 3;               // 0..3
  const int arow  = rt * 16 + (lane & 15);
  const int khalf = (lane >> 4) << 3;      // A: lanes>=16 hold K+8 block

  v8f zero = {0.f,0.f,0.f,0.f,0.f,0.f,0.f,0.f};
  v8f acc[4];
#pragma unroll
  for (int j = 0; j < 4; ++j) acc[j] = zero;

  for (int kt = 0; kt < 8; ++kt) {
    const int kb = kt * 32 + khalf;
    const float* ap = A + arow * 256 + kb;
    float4 a0 = *(const float4*)(ap);
    float4 a1 = *(const float4*)(ap + 4);
    float4 a2 = *(const float4*)(ap + 16);
    float4 a3 = *(const float4*)(ap + 20);
    BF16Vec av;
    av.u[0]=f2bf(a0.x);  av.u[1]=f2bf(a0.y);  av.u[2]=f2bf(a0.z);  av.u[3]=f2bf(a0.w);
    av.u[4]=f2bf(a1.x);  av.u[5]=f2bf(a1.y);  av.u[6]=f2bf(a1.z);  av.u[7]=f2bf(a1.w);
    av.u[8]=f2bf(a2.x);  av.u[9]=f2bf(a2.y);  av.u[10]=f2bf(a2.z); av.u[11]=f2bf(a2.w);
    av.u[12]=f2bf(a3.x); av.u[13]=f2bf(a3.y); av.u[14]=f2bf(a3.z); av.u[15]=f2bf(a3.w);
#pragma unroll
    for (int j = 0; j < 4; ++j) {
      const int ct = cg * 4 + j;
      const v8us* bp = (const v8us*)(Wp + (((kt * 16 + ct) * 32 + lane) << 4));
      v8us b0 = bp[0], b1 = bp[1];
      BF16Vec bm;
#pragma unroll
      for (int e = 0; e < 8; ++e) { bm.u[e] = b0[e]; bm.u[e + 8] = b1[e]; }
      acc[j] = __builtin_amdgcn_wmma_f32_16x16x32_bf16(
          false, av.b, false, bm.b, (short)0, acc[j], false, false);
    }
  }
#pragma unroll
  for (int j = 0; j < 4; ++j) {
    const int c = (cg * 4 + j) * 16 + (lane & 15);
    const float bb = bias[c];
#pragma unroll
    for (int r = 0; r < 8; ++r) {
      const int m = rt * 16 + r + khalf;   // C/D layout: lanes>=16 -> M+8
      Out[m * 256 + c] = fmaxf(acc[j][r] + bb, 0.f);
    }
  }
}

// ------------------------------------------------------------------
// Fused per-(b,n) kernel
// ------------------------------------------------------------------
#define ES  264   // enc LDS row stride (halfs)
#define KVS 264   // kh/vh LDS row stride (floats)

__device__ __forceinline__ void tile_gemm_bias_relu(
    const unsigned short* s_enc, float* s_kv,
    const unsigned short* __restrict__ Wp, const float* __restrict__ bias,
    int w, int lane) {
  const int mt     = w & 3;                // row tile (t-dim, 4 tiles of 16)
  const int ctbase = (w >> 2) * 8;         // 8 col tiles per wave
  const int arow   = mt * 16 + (lane & 15);
  const int khalf  = (lane >> 4) << 3;

  v8f zero = {0.f,0.f,0.f,0.f,0.f,0.f,0.f,0.f};
  v8f acc[8];
#pragma unroll
  for (int j = 0; j < 8; ++j) acc[j] = zero;

  for (int kt = 0; kt < 8; ++kt) {
    const int kb = kt * 32 + khalf;
    v8us a0 = *(const v8us*)(s_enc + arow * ES + kb);
    v8us a1 = *(const v8us*)(s_enc + arow * ES + kb + 16);
    BF16Vec av;
#pragma unroll
    for (int e = 0; e < 8; ++e) { av.u[e] = a0[e]; av.u[e + 8] = a1[e]; }
#pragma unroll
    for (int j = 0; j < 8; ++j) {
      const v8us* bp = (const v8us*)(Wp + (((kt * 16 + ctbase + j) * 32 + lane) << 4));
      v8us b0 = bp[0], b1 = bp[1];
      BF16Vec bm;
#pragma unroll
      for (int e = 0; e < 8; ++e) { bm.u[e] = b0[e]; bm.u[e + 8] = b1[e]; }
      acc[j] = __builtin_amdgcn_wmma_f32_16x16x32_bf16(
          false, av.b, false, bm.b, (short)0, acc[j], false, false);
    }
  }
#pragma unroll
  for (int j = 0; j < 8; ++j) {
    const int c = (ctbase + j) * 16 + (lane & 15);
    const float bb = bias[c];
#pragma unroll
    for (int r = 0; r < 8; ++r) {
      const int m = mt * 16 + r + khalf;
      if (m < HIS_) s_kv[m * KVS + c] = fmaxf(acc[j][r] + bb, 0.f);
    }
  }
}

__global__ __launch_bounds__(256) void fused_attn_kernel(
    const float* __restrict__ Kin,    // [B,HIS,N,DK]
    const float* __restrict__ bk, const float* __restrict__ bv,
    const float* __restrict__ Wenc,   // [DK,256]
    const int* __restrict__ idx, const float* __restrict__ pe,
    const float* __restrict__ qh,     // [2048,256] (post-relu)
    const unsigned short* __restrict__ Wkp, const unsigned short* __restrict__ Wvp,
    float* __restrict__ outvec) {     // [2048,256]
  __shared__ __attribute__((aligned(16))) unsigned short s_enc[64 * ES];
  __shared__ __attribute__((aligned(16))) float s_kv[HIS_ * KVS];
  __shared__ float s_neigh[HIS_ * 12];
  __shared__ float s_qh[256];
  __shared__ float s_att[NV_ * 64];

  const int tid  = threadIdx.x;
  const int w    = tid >> 5;
  const int lane = tid & 31;
  const int n    = blockIdx.x;
  const int b    = blockIdx.y;
  const int bid  = b * N_AG + n;

  // ---- phase 0: load qh row, gather neighbor actions ----
  s_qh[tid] = qh[bid * 256 + tid];
  if (tid < HIS_ * DK_) {
    int t = tid / DK_, d = tid - t * DK_;
    int a = idx[n * HIS_ + t];
    s_neigh[t * 12 + d] = Kin[((b * HIS_ + t) * N_AG + a) * DK_ + d];
  }
  __syncthreads();

  // ---- phase 1: enc[t][e] = gelu(neigh . Wenc) + pe, stored bf16 ----
  {
    float wreg[DK_];
#pragma unroll
    for (int d = 0; d < DK_; ++d) wreg[d] = Wenc[d * 256 + tid];
    for (int t = 0; t < HIS_; ++t) {
      float x = 0.f;
#pragma unroll
      for (int d = 0; d < DK_; ++d) x += s_neigh[t * 12 + d] * wreg[d];
      float g = 0.5f * x * (1.0f + erff(x * 0.70710678118654752f));  // exact GELU
      s_enc[t * ES + tid] = f2bf(g + pe[t * 256 + tid]);
    }
    for (int t = HIS_; t < 64; ++t) s_enc[t * ES + tid] = 0;  // zero-pad M to 64
  }
  __syncthreads();

  // ---- phase 2: kh = relu(enc @ Wk + bk) via WMMA ----
  tile_gemm_bias_relu(s_enc, s_kv, Wkp, bk, w, lane);
  __syncthreads();

  // ---- phase 3: attention logits + softmax (wave w == head v) ----
  {
    const int v = w;
    const float* qv = s_qh + v * DV_;
    const int t0 = lane, t1 = lane + 32;
    const bool ok1 = (t1 < HIS_);
    float d0 = 0.f, d1 = 0.f;
#pragma unroll 8
    for (int j = 0; j < DV_; ++j) {
      float qj = qv[j];
      d0 += qj * s_kv[t0 * KVS + v * DV_ + j];
      if (ok1) d1 += qj * s_kv[t1 * KVS + v * DV_ + j];
    }
    float l0 = d0 * (1.0f / 3.0f);                 // / sqrt(DK=9)
    float l1 = ok1 ? d1 * (1.0f / 3.0f) : -1e30f;
    float mx = fmaxf(l0, l1);
    for (int o = 16; o > 0; o >>= 1) mx = fmaxf(mx, __shfl_xor(mx, o));
    float e0 = __expf(l0 - mx);
    float e1 = ok1 ? __expf(l1 - mx) : 0.f;
    float sm = e0 + e1;
    for (int o = 16; o > 0; o >>= 1) sm += __shfl_xor(sm, o);
    float inv = 1.0f / sm;
    s_att[v * 64 + t0] = e0 * inv;
    s_att[v * 64 + t1] = e1 * inv;
  }
  __syncthreads();

  // ---- phase 4: vh = relu(enc @ Wv + bv), reusing s_kv ----
  tile_gemm_bias_relu(s_enc, s_kv, Wvp, bv, w, lane);
  __syncthreads();

  // ---- phase 5: out[c] = sum_t att[v][t] * vh[t][c], c = tid = v*32+d ----
  {
    const int v = tid >> 5;
    const float* av = s_att + v * 64;
    float acc = 0.f;
    for (int t = 0; t < HIS_; ++t) acc += av[t] * s_kv[t * KVS + tid];
    outvec[bid * 256 + tid] = acc;
  }
}

// ------------------------------------------------------------------
// Host launcher
// ------------------------------------------------------------------
extern "C" void kernel_launch(void* const* d_in, const int* in_sizes, int n_in,
                              void* d_out, int out_size, void* d_ws, size_t ws_size,
                              hipStream_t stream) {
  const float* q    = (const float*)d_in[0];
  const float* k    = (const float*)d_in[1];
  const float* adjs = (const float*)d_in[2];
  const float* Wq   = (const float*)d_in[3];
  const float* bq   = (const float*)d_in[4];
  const float* Wk   = (const float*)d_in[5];
  const float* bk   = (const float*)d_in[6];
  const float* Wv   = (const float*)d_in[7];
  const float* bv   = (const float*)d_in[8];
  const float* Wout = (const float*)d_in[9];
  const float* bout = (const float*)d_in[10];
  const float* Wenc = (const float*)d_in[11];
  float* out = (float*)d_out;
  (void)in_sizes; (void)n_in; (void)out_size; (void)ws_size;

  char* ws = (char*)d_ws;
  unsigned short* wpack = (unsigned short*)ws;      // 524288 B (4 packed matrices)
  unsigned short* wqp = wpack;
  unsigned short* wkp = wpack + 65536;
  unsigned short* wvp = wpack + 131072;
  unsigned short* wop = wpack + 196608;
  int*   idx = (int*)  (ws + 524288);               // 12800 B
  float* pe  = (float*)(ws + 537088);               // 51200 B
  float* qh  = (float*)(ws + 588288);               // 2 MB
  float* ov  = (float*)(ws + 588288 + 2097152);     // 2 MB

  const int prep_items = 262144 + N_AG * HIS_ + HIS_ * 256;
  prep_kernel<<<(prep_items + 255) / 256, 256, 0, stream>>>(
      Wq, Wk, Wv, Wout, adjs, wpack, idx, pe);

  gemm256_relu_kernel<<<64, 256, 0, stream>>>(q, wqp, bq, qh);

  fused_attn_kernel<<<dim3(N_AG, B_SZ), 256, 0, stream>>>(
      k, bk, bv, Wenc, idx, pe, qh, wkp, wvp, ov);

  gemm256_relu_kernel<<<64, 256, 0, stream>>>(ov, wop, bout, out);
}
